// GCN_DECONF_17411797418342
// MI455X (gfx1250) — matrix operand order repaired and checked by reading, not compiled, not run on hardware
//
#include <hip/hip_runtime.h>
#include <hip/hip_bf16.h>

// ---------------------------------------------------------------------------
// GCN_DECONF on MI455X (gfx1250): 2x GCNConv + 2 MLP heads + scalar heads.
// Dense GEMMs via v_wmma_f32_16x16x32_f16 (f32 accumulate), LDS tiles staged
// with the Tensor Data Mover (tensor_load_to_lds + s_wait_tensorcnt) when the
// toolchain exposes it, edge aggregation via L2-resident global_atomic_add_f32,
// wave32 shuffles for row reductions.
// ---------------------------------------------------------------------------

typedef __attribute__((ext_vector_type(16))) _Float16 v16h;
typedef __attribute__((ext_vector_type(8)))  _Float16 v8h;
typedef __attribute__((ext_vector_type(4)))  _Float16 v4h;
typedef __attribute__((ext_vector_type(8)))  float    v8f;

#define HID 128

#if defined(__HIP_DEVICE_COMPILE__) && \
    __has_builtin(__builtin_amdgcn_tensor_load_to_lds) && \
    __has_builtin(__builtin_amdgcn_s_wait_tensorcnt)
#define USE_TDM 1
#else
#define USE_TDM 0
#endif

#if USE_TDM
typedef __attribute__((ext_vector_type(4))) unsigned int tdm_v4u;
typedef __attribute__((ext_vector_type(8))) int          tdm_v8i;
typedef __attribute__((ext_vector_type(4))) int          tdm_v4i;

// DMA one contiguous 32KB (16384 f16) block global -> LDS via the TDM.
// D# per CDNA5 ISA ch.8: group0 = {count=1 | lds_addr | global_addr | type=2},
// group1 = {data_size=2B, tensor_dim0=tile_dim0=16384, dim1=1, stride0=16384}.
__device__ __forceinline__ void tdm_copy_32k(unsigned long long gaddr,
                                             unsigned int lds_byte_off) {
  const unsigned int NE = 16384;  // elements (f16)
  tdm_v4u g0;
  g0[0] = 1u;                                                   // count=1 (user D#)
  g0[1] = lds_byte_off;                                         // lds_addr
  g0[2] = (unsigned int)(gaddr & 0xFFFFFFFFull);                // global_addr[31:0]
  g0[3] = (unsigned int)((gaddr >> 32) & 0x01FFFFFFull)         // global_addr[56:32]
          | (2u << 30);                                         // type=2 ("image")
  tdm_v8i g1;
  g1[0] = 0x00010000;                    // workgroup_mask=0, data_size=1 (2 bytes)
  g1[1] = (int)(NE << 16);               // tensor_dim0[15:0]   @ bits 63:48
  g1[2] = (int)((NE >> 16) | (1u << 16));// tensor_dim0[31:16], tensor_dim1=1
  g1[3] = (int)(NE << 16);               // tile_dim0           @ bits 127:112
  g1[4] = 1;                             // tile_dim1=1, tile_dim2=0
  g1[5] = (int)NE;                       // tensor_dim0_stride[31:0]
  g1[6] = 0;                             // stride0 hi, tensor_dim1_stride lo
  g1[7] = 0;
  tdm_v4i gz = {0, 0, 0, 0};             // groups 2/3 unused (<=2D tensor)
#if __clang_major__ >= 23
  tdm_v8i gz8 = {0, 0, 0, 0, 0, 0, 0, 0};
  __builtin_amdgcn_tensor_load_to_lds(g0, g1, gz, gz, gz8, 0);
#else
  __builtin_amdgcn_tensor_load_to_lds(g0, g1, gz, gz, 0);
#endif
}
#endif  // USE_TDM

// ---------- small elementwise kernels ----------

__global__ __launch_bounds__(256) void zero_f32(float* __restrict__ p, int n) {
  int i = blockIdx.x * blockDim.x + threadIdx.x;
  if (i < n) p[i] = 0.0f;
}

__global__ __launch_bounds__(256) void convert_f16(const float* __restrict__ src,
                                                   _Float16* __restrict__ dst, int n) {
  int i = blockIdx.x * blockDim.x + threadIdx.x;
  if (i < n) dst[i] = (_Float16)src[i];
}

// transpose+convert the 4 [128,128] weight mats into f16 Wt[n][k] = W[k][n]
__global__ __launch_bounds__(256) void prep_weights(const float* __restrict__ W0,
                                                    const float* __restrict__ W1,
                                                    const float* __restrict__ Wa,
                                                    const float* __restrict__ Wb,
                                                    _Float16* __restrict__ Wt) {
  int idx = blockIdx.x * blockDim.x + threadIdx.x;     // 0 .. 4*128*128-1
  int which = idx >> 14;
  int rem   = idx & 16383;
  int k = rem >> 7;
  int n = rem & 127;
  const float* src = (which == 0) ? W0 : (which == 1) ? W1 : (which == 2) ? Wa : Wb;
  Wt[which * 16384 + n * HID + k] = (_Float16)src[k * HID + n];
}

__global__ __launch_bounds__(256) void count_deg(const int* __restrict__ col,
                                                 float* __restrict__ deg, int ne) {
  int i = blockIdx.x * blockDim.x + threadIdx.x;
  if (i < ne)
    __hip_atomic_fetch_add(&deg[col[i]], 1.0f, __ATOMIC_RELAXED, __HIP_MEMORY_SCOPE_AGENT);
}

__global__ __launch_bounds__(256) void compute_dinv(const float* __restrict__ deg,
                                                    float* __restrict__ dinv, int n) {
  int i = blockIdx.x * blockDim.x + threadIdx.x;
  if (i < n) dinv[i] = rsqrtf(deg[i] + 1.0f);
}

// agg[i][c] = h[i][c] * dinv[i]^2   (self-loop term, also initializes agg)
__global__ __launch_bounds__(256) void init_agg(const float* __restrict__ h,
                                                const float* __restrict__ dinv,
                                                float* __restrict__ agg, int total) {
  int i = blockIdx.x * blockDim.x + threadIdx.x;
  if (i < total) {
    float d = dinv[i >> 7];
    agg[i] = h[i] * d * d;
  }
}

// relu(agg + b) -> f16 activation buffer for the next WMMA GEMM
__global__ __launch_bounds__(256) void finalize_relu_f16(const float* __restrict__ agg,
                                                         const float* __restrict__ b,
                                                         _Float16* __restrict__ outh,
                                                         int total) {
  int i = blockIdx.x * blockDim.x + threadIdx.x;
  if (i < total) {
    float v = agg[i] + b[i & 127];
    outh[i] = (_Float16)fmaxf(v, 0.0f);
  }
}

// ---------- edge scatter-add: one wave32 per edge, lane covers 4 columns ----------

__global__ __launch_bounds__(256) void edge_agg(const int* __restrict__ rows,
                                                const int* __restrict__ cols,
                                                const float* __restrict__ dinv,
                                                const float* __restrict__ h,
                                                float* __restrict__ agg, int ne) {
  int wid   = (blockIdx.x * blockDim.x + threadIdx.x) >> 5;
  int lane  = threadIdx.x & 31;
  int nwave = (gridDim.x * blockDim.x) >> 5;
  for (int e = wid; e < ne; e += nwave) {
    int r = rows[e], c = cols[e];
    float norm = dinv[r] * dinv[c];
    float4 hv = ((const float4*)(h + (size_t)r * HID))[lane];
    float* d = agg + (size_t)c * HID + lane * 4;
    __hip_atomic_fetch_add(d + 0, hv.x * norm, __ATOMIC_RELAXED, __HIP_MEMORY_SCOPE_AGENT);
    __hip_atomic_fetch_add(d + 1, hv.y * norm, __ATOMIC_RELAXED, __HIP_MEMORY_SCOPE_AGENT);
    __hip_atomic_fetch_add(d + 2, hv.z * norm, __ATOMIC_RELAXED, __HIP_MEMORY_SCOPE_AGENT);
    __hip_atomic_fetch_add(d + 3, hv.w * norm, __ATOMIC_RELAXED, __HIP_MEMORY_SCOPE_AGENT);
  }
}

// ---------- WMMA GEMM: Out[N,128] = Ah[N,128] @ W  (W given transposed f16) ----------
// Block = 256 threads = 8 waves; block tile = 128 rows. Wave w owns rows w*16..+15,
// all 128 cols: 8 accumulators (16x16), K=128 in 4 steps of 32.

__global__ __launch_bounds__(256) void gemm128_f16(const _Float16* __restrict__ Ah,
                                                   const _Float16* __restrict__ Wt,
                                                   float* __restrict__ Out, int nrows) {
  // NOTE: declared in this order so sA sits at LDS offset 0, sB at 32768 (the
  // TDM descriptor uses these raw LDS byte offsets).
  __shared__ _Float16 sA[HID * HID];   // 32 KB activation tile (row-major)
  __shared__ _Float16 sB[HID * HID];   // 32 KB Wt[n][k]
  const int tid = threadIdx.x;
  const size_t rowBase = (size_t)blockIdx.x * HID;

#if USE_TDM
  if (tid < 32) {  // wave 0 drives both DMAs; TENSORcnt orders them in-issue-order
    tdm_copy_32k((unsigned long long)(size_t)(Ah + rowBase * HID), 0u);
    tdm_copy_32k((unsigned long long)(size_t)Wt, (unsigned int)(HID * HID * 2));
    __builtin_amdgcn_s_wait_tensorcnt(0);
  }
  __syncthreads();
  // The TDM builtin only carries opaque descriptor words, so the compiler
  // cannot see that it wrote sA/sB. Escape the LDS pointers into an opaque
  // asm with a memory clobber so the fragment loads below are not treated as
  // loads of never-written (undef) shared memory.
  asm volatile("" : : "v"(&sA[0]), "v"(&sB[0]) : "memory");
#else
  {  // cooperative 16B staging (coalesced global_load_b128 -> ds_store_b128)
    const uint4* gA = (const uint4*)(Ah + rowBase * HID);
    const uint4* gB = (const uint4*)Wt;
    uint4* lA = (uint4*)sA;
    uint4* lB = (uint4*)sB;
#pragma unroll
    for (int i = 0; i < 8; ++i) {
      lA[tid + i * 256] = gA[tid + i * 256];
      lB[tid + i * 256] = gB[tid + i * 256];
    }
  }
  __syncthreads();
#endif

  const int wave = tid >> 5;
  const int lane = tid & 31;
  const int g    = lane >> 4;   // lane group 0/1
  const int lm   = lane & 15;
  const int m0   = wave * 16;

  v8f acc[8];
#pragma unroll
  for (int n = 0; n < 8; ++n) acc[n] = (v8f){0.f, 0.f, 0.f, 0.f, 0.f, 0.f, 0.f, 0.f};

#pragma unroll
  for (int t = 0; t < 4; ++t) {
    // A fragment (16x32 f16): lane holds row m0+lm, K = 32t+16g+[0..15]
    const _Float16* arow = sA + (m0 + lm) * HID + t * 32 + 16 * g;
    v8h a_lo = *(const v8h*)(arow);
    v8h a_hi = *(const v8h*)(arow + 8);
    v16h afrag;
#pragma unroll
    for (int i = 0; i < 8; ++i) { afrag[i] = a_lo[i]; afrag[i + 8] = a_hi[i]; }

    // Batch all 8 B-fragment loads (contiguous 32B v16h each), then burst the
    // WMMAs so the DS waits don't serialize each matrix op.
    v16h bf[8];
#pragma unroll
    for (int n = 0; n < 8; ++n)
      bf[n] = *(const v16h*)(sB + (n * 16 + lm) * HID + t * 32 + 16 * g);
#pragma unroll
    for (int n = 0; n < 8; ++n)
      acc[n] = __builtin_amdgcn_wmma_f32_16x16x32_f16(
          false, afrag, false, bf[n], (short)0, acc[n], false, false);
  }

  // C/D layout: VGPR v -> M = v + 8g, N = lm
#pragma unroll
  for (int n = 0; n < 8; ++n) {
#pragma unroll
    for (int v = 0; v < 8; ++v) {
      size_t grow = rowBase + (size_t)(m0 + v + g * 8);
      if (grow < (size_t)nrows) Out[grow * HID + n * 16 + lm] = acc[n][v];
    }
  }
}

// ---------- per-row heads: y = dot(relu(H + b), w) + b_out ----------

__global__ __launch_bounds__(256) void head_dot(const float* __restrict__ H,
                                                const float* __restrict__ bias,
                                                const float* __restrict__ w,
                                                const float* __restrict__ bout,
                                                float* __restrict__ out, int nrows) {
  int wid  = (blockIdx.x * blockDim.x + threadIdx.x) >> 5;
  int lane = threadIdx.x & 31;
  if (wid >= nrows) return;
  float4 hv = ((const float4*)(H + (size_t)wid * HID))[lane];
  float4 bv = ((const float4*)bias)[lane];
  float4 wv = ((const float4*)w)[lane];
  float s = fmaxf(hv.x + bv.x, 0.f) * wv.x + fmaxf(hv.y + bv.y, 0.f) * wv.y +
            fmaxf(hv.z + bv.z, 0.f) * wv.z + fmaxf(hv.w + bv.w, 0.f) * wv.w;
#pragma unroll
  for (int off = 16; off > 0; off >>= 1) s += __shfl_xor(s, off, 32);
  if (lane == 0) out[wid] = s + bout[0];
}

// ---------- final: p1 = sigmoid(rep.pp_w + pp_b); y = t>0 ? y1 : y0 ----------

__global__ __launch_bounds__(256) void final_out(const _Float16* __restrict__ reph,
                                                 const float* __restrict__ ppw,
                                                 const float* __restrict__ ppb,
                                                 const int* __restrict__ t,
                                                 const float* __restrict__ y0,
                                                 const float* __restrict__ y1,
                                                 float* __restrict__ out, int nrows) {
  int wid  = (blockIdx.x * blockDim.x + threadIdx.x) >> 5;
  int lane = threadIdx.x & 31;
  if (wid >= nrows) return;
  v4h rv = *(const v4h*)(reph + (size_t)wid * HID + lane * 4);
  float4 wv = ((const float4*)ppw)[lane];
  float s = (float)rv[0] * wv.x + (float)rv[1] * wv.y +
            (float)rv[2] * wv.z + (float)rv[3] * wv.w;
#pragma unroll
  for (int off = 16; off > 0; off >>= 1) s += __shfl_xor(s, off, 32);
  if (lane == 0) {
    float p = 1.0f / (1.0f + __expf(-(s + ppb[0])));
    out[wid] = p;
    out[nrows + wid] = (t[wid] > 0) ? y1[wid] : y0[wid];
  }
}

// ---------------------------------------------------------------------------

extern "C" void kernel_launch(void* const* d_in, const int* in_sizes, int n_in,
                              void* d_out, int out_size, void* d_ws, size_t ws_size,
                              hipStream_t stream) {
  const float* x   = (const float*)d_in[0];
  const int*   t   = (const int*)d_in[1];
  const int*   ei  = (const int*)d_in[3];
  const float* W0  = (const float*)d_in[4];
  const float* b0  = (const float*)d_in[5];
  const float* W1  = (const float*)d_in[6];
  const float* b1  = (const float*)d_in[7];
  const float* W00 = (const float*)d_in[8];
  const float* b00 = (const float*)d_in[9];
  const float* W10 = (const float*)d_in[10];
  const float* b10 = (const float*)d_in[11];
  const float* w01 = (const float*)d_in[12];
  const float* b01 = (const float*)d_in[13];
  const float* w11 = (const float*)d_in[14];
  const float* b11 = (const float*)d_in[15];
  const float* ppw = (const float*)d_in[16];
  const float* ppb = (const float*)d_in[17];

  const int n  = in_sizes[0] / HID;   // 100000
  const int ne = in_sizes[3] / 2;     // 1600000
  float* out = (float*)d_out;
  (void)n_in; (void)out_size; (void)ws_size;

  // carve workspace (ah deliberately NOT last: last GEMM block over-reads a bit)
  char* p = (char*)d_ws;
  auto carve = [&](size_t bytes) {
    char* r = p;
    p += (bytes + 255) & ~(size_t)255;
    return r;
  };
  _Float16* Wt   = (_Float16*)carve((size_t)4 * 128 * 128 * sizeof(_Float16));
  float*    deg  = (float*)carve((size_t)n * 4);
  float*    dinv = (float*)carve((size_t)n * 4);
  _Float16* ah   = (_Float16*)carve((size_t)n * HID * 2);
  float*    hbuf = (float*)carve((size_t)n * HID * 4);
  float*    agg  = (float*)carve((size_t)n * HID * 4);
  float*    y0b  = (float*)carve((size_t)n * 4);
  float*    y1b  = (float*)carve((size_t)n * 4);

  const int TPB = 256;
  const int total   = n * HID;
  const int gElem   = (total + TPB - 1) / TPB;
  const int gNode   = (n + TPB - 1) / TPB;
  const int gEdge   = (ne + TPB - 1) / TPB;
  const int gGemm   = (n + 127) / 128;
  const int gWave   = (n * 32 + TPB - 1) / TPB;   // one wave32 per node

  prep_weights<<<(4 * 128 * 128) / TPB, TPB, 0, stream>>>(
      W0, W1, W00 + 128 * 128, W10 + 128 * 128, Wt);
  zero_f32<<<gNode, TPB, 0, stream>>>(deg, n);
  convert_f16<<<gElem, TPB, 0, stream>>>(x, ah, total);
  count_deg<<<gEdge, TPB, 0, stream>>>(ei + ne, deg, ne);
  compute_dinv<<<gNode, TPB, 0, stream>>>(deg, dinv, n);

  // ---- GCN layer 0 ----
  gemm128_f16<<<gGemm, TPB, 0, stream>>>(ah, Wt + 0 * 16384, hbuf, n);
  init_agg<<<gElem, TPB, 0, stream>>>(hbuf, dinv, agg, total);
  edge_agg<<<2048, TPB, 0, stream>>>(ei, ei + ne, dinv, hbuf, agg, ne);
  finalize_relu_f16<<<gElem, TPB, 0, stream>>>(agg, b0, ah, total);

  // ---- GCN layer 1 ----
  gemm128_f16<<<gGemm, TPB, 0, stream>>>(ah, Wt + 1 * 16384, hbuf, n);
  init_agg<<<gElem, TPB, 0, stream>>>(hbuf, dinv, agg, total);
  edge_agg<<<2048, TPB, 0, stream>>>(ei, ei + ne, dinv, hbuf, agg, ne);
  finalize_relu_f16<<<gElem, TPB, 0, stream>>>(agg, b1, ah, total);

  // ---- head 0 ----
  gemm128_f16<<<gGemm, TPB, 0, stream>>>(ah, Wt + 2 * 16384, hbuf, n);
  head_dot<<<gWave, TPB, 0, stream>>>(hbuf, b00 + 128, w01, b01, y0b, n);

  // ---- head 1 ----
  gemm128_f16<<<gGemm, TPB, 0, stream>>>(ah, Wt + 3 * 16384, hbuf, n);
  head_dot<<<gWave, TPB, 0, stream>>>(hbuf, b10 + 128, w11, b11, y1b, n);

  // ---- propensity + select ----
  final_out<<<gWave, TPB, 0, stream>>>(ah, ppw, ppb, t, y0b, y1b, out, n);
}
